// Decoder_31988916421097
// MI455X (gfx1250) — compile-verified, matrix operand order checked by reading, not run on hardware
//
#include <hip/hip_runtime.h>

// ---------------------------------------------------------------------------
// Decoder: LSTM + dual cross-attention on MI455X (gfx1250).
// v3: all GEMM operands pre-converted to bf16 once; GEMM/attention inner
// loops are pure  global_load_async_to_lds_b128 -> ds_load_b128 -> v_wmma.
// The serial LSTM-step GEMM (only 32 blocks -> no occupancy-based latency
// hiding) now double-buffers its LDS tiles and pipelines the async DMA one
// tile ahead with a non-zero  s_wait_asynccnt  (in-order per-wave ASYNCcnt).
// ---------------------------------------------------------------------------

#define BSZ   16
#define TDEC  256
#define TSRC  512
#define EMB   1024
#define HID   1024
#define GATES 4096   // 4*HID, torch gate order i,f,g,o
#define NHEAD 16
#define DHEAD 64

typedef __attribute__((ext_vector_type(16))) __bf16          bf16x16;
typedef __attribute__((ext_vector_type(8)))  float           floatx8;
typedef __attribute__((ext_vector_type(8)))  unsigned short  ushortx8;

union FragCvt { bf16x16 v; ushortx8 h[2]; };

__device__ __forceinline__ unsigned short f2bf(float f) {
  unsigned u = __builtin_bit_cast(unsigned, f);
  u += 0x7FFFu + ((u >> 16) & 1u);            // round-to-nearest-even
  return (unsigned short)(u >> 16);
}

// ISA 10.2: generic LDS addresses are {SHARED_BASE, offset[31:0]} -> the low
// 32 bits of a generic pointer to __shared__ memory are the LDS byte offset
// that GLOBAL_LOAD_ASYNC_TO_LDS expects in its VDST VGPR.
__device__ __forceinline__ unsigned lds_off_u32(const void* p) {
  return (unsigned)(unsigned long long)p;
}

// Async DMA of one 16B chunk per lane: global -> LDS, tracked by ASYNCcnt.
__device__ __forceinline__ void async_b128(unsigned lds_off, const void* gaddr) {
  asm volatile("global_load_async_to_lds_b128 %0, %1, off"
               :: "v"(lds_off), "v"(gaddr) : "memory");
}
__device__ __forceinline__ void wait_async0() {
  asm volatile("s_wait_asynccnt 0" ::: "memory");
}
template <int N>
__device__ __forceinline__ void wait_asyncN() {
  asm volatile("s_wait_asynccnt %0" :: "i"(N) : "memory");
}

// 16-bit WMMA operand fragment from LDS tile [16 rows][stride] (ISA 7.12.2):
// lane&15 = row, lane>>4 = K-half; two aligned 16B LDS reads per fragment.
__device__ __forceinline__ bf16x16 load_frag(const unsigned short* base, int stride) {
  const int lane = threadIdx.x & 31;
  const int r    = lane & 15;
  const int kg   = (lane >> 4) << 3;          // 0 or 8
  FragCvt f;
  f.h[0] = *(const ushortx8*)(base + r * stride + kg);
  f.h[1] = *(const ushortx8*)(base + r * stride + kg + 16);
  return f.v;
}

__device__ __forceinline__ floatx8 wmma_bf16(bf16x16 a, bf16x16 b, floatx8 c) {
  return __builtin_amdgcn_wmma_f32_16x16x32_bf16(false, a, false, b, (short)0, c,
                                                 false, false);
}

// ---------------------------------------------------------------------------
// One-shot f32 -> bf16 conversion, 8 elements/thread (n multiple of 2048).
// ---------------------------------------------------------------------------
__global__ __launch_bounds__(256) void cvt_f32_bf16(const float* __restrict__ src,
                                                    unsigned short* __restrict__ dst)
{
  const size_t i = ((size_t)blockIdx.x * 256 + threadIdx.x) * 8;
  __builtin_prefetch(src + i + 2048 * 8, 0, 1);   // global_prefetch_b8
  const float4 a = *(const float4*)(src + i);
  const float4 b = *(const float4*)(src + i + 4);
  ushortx8 o;
  o[0] = f2bf(a.x); o[1] = f2bf(a.y); o[2] = f2bf(a.z); o[3] = f2bf(a.w);
  o[4] = f2bf(b.x); o[5] = f2bf(b.y); o[6] = f2bf(b.z); o[7] = f2bf(b.w);
  *(ushortx8*)(dst + i) = o;
}

// ---------------------------------------------------------------------------
// GEMM: C[M,N] = A[M,K](bf16) * W[N,K]^T(bf16) + bias, optional f32 residual,
// output f32 (strided, for the [B,T,3H] concat) or bf16.
// 256 thr = 8 waves; BM=32, BN=256, BK=64; latency hidden by block
// oversubscription (grids of 512..2048 blocks), so single-buffered LDS.
// ---------------------------------------------------------------------------
__global__ __launch_bounds__(256) void gemm_bf(
    const unsigned short* __restrict__ A, const unsigned short* __restrict__ W,
    const float* __restrict__ bias, const float* __restrict__ residual, int ldr,
    void* __restrict__ Cout, int ldc, int c_is_bf16, int K)
{
  __shared__ __attribute__((aligned(16))) unsigned short As[32 * 64];   //  4KB
  __shared__ __attribute__((aligned(16))) unsigned short Bs[256 * 64];  // 32KB
  const int tid = threadIdx.x;
  const int w   = tid >> 5;
  const int wm  = w & 1;
  const int wn  = w >> 1;
  const int bm  = blockIdx.y;
  const int bn  = blockIdx.x;

  floatx8 acc[4] = {};

  const int ar = tid >> 3;               // A tile row 0..31
  const int ac = (tid & 7) << 3;         // A tile col chunk
  const unsigned aoff = lds_off_u32(As + ar * 64 + ac);
  const unsigned boff = lds_off_u32(Bs + tid * 64);
  const unsigned short* arow = A + (size_t)(bm * 32 + ar) * K + ac;
  const unsigned short* wrow = W + (size_t)(bn * 256 + tid) * K;

  for (int k0 = 0; k0 < K; k0 += 64) {
    __syncthreads();                     // prior fragment reads done
    async_b128(aoff, arow + k0);         // A tile: 1 chunk/thread
    for (int j = 0; j < 64; j += 8)      // B tile: one row/thread, 8 chunks
      async_b128(boff + j * 2, wrow + k0 + j);
    wait_async0();
    __syncthreads();                     // all waves' DMA landed
    for (int kk = 0; kk < 64; kk += 32) {
      const bf16x16 afrag = load_frag(As + wm * 16 * 64 + kk, 64);
      for (int i = 0; i < 4; ++i) {
        const bf16x16 bfrag = load_frag(Bs + (wn * 64 + i * 16) * 64 + kk, 64);
        acc[i] = wmma_bf16(afrag, bfrag, acc[i]);
      }
    }
  }

  // C/D layout: lane&15 = N col, VGPR r = M row r + 8*(lane>>4).
  const int lane = tid & 31;
  const int n  = lane & 15;
  const int mb = (lane >> 4) << 3;
  for (int i = 0; i < 4; ++i) {
    const int col = bn * 256 + wn * 64 + i * 16 + n;
    const float bv = bias ? bias[col] : 0.0f;
    for (int r = 0; r < 8; ++r) {
      const int row = bm * 32 + wm * 16 + mb + r;
      float v = acc[i][r] + bv;
      if (residual) v += residual[(size_t)row * ldr + col];
      if (c_is_bf16)
        ((unsigned short*)Cout)[(size_t)row * (size_t)ldc + col] = f2bf(v);
      else
        ((float*)Cout)[(size_t)row * (size_t)ldc + col] = v;
    }
  }
}

// ---------------------------------------------------------------------------
// LSTM step GEMM: gates[16,4096] = h_prev(bf16) @ W_hh^T(bf16) + b_hh + x_proj
// Critical serial path (32 blocks/step, 256 steps): double-buffered LDS tiles
// with async DMA issued one tile ahead. Min per-wave issues per tile is 4
// (B chunks; A adds a 5th on waves 0-3 when t>0) and ASYNCcnt retires in
// order, so  s_wait_asynccnt 4  after issuing tile i+1 guarantees tile i
// landed while tile i+1's DMA overlaps the WMMAs.
// ---------------------------------------------------------------------------
__global__ __launch_bounds__(256) void lstm_step_gemm(
    const unsigned short* __restrict__ hprev,   // bf16 [16][1024]
    const unsigned short* __restrict__ Whh,     // bf16 [4096][1024]
    const float* __restrict__ bhh,
    const float* __restrict__ xproj,            // f32 [B*T][4096]
    int t, float* __restrict__ gates)           // f32 [16][4096]
{
  __shared__ __attribute__((aligned(16))) unsigned short As[2][16 * 64];   //  4KB
  __shared__ __attribute__((aligned(16))) unsigned short Bs[2][128 * 64];  // 32KB
  const int tid = threadIdx.x;
  const int w   = tid >> 5;
  const int bn  = blockIdx.x;                   // 0..31
  floatx8 acc = {};

  const int ar = tid >> 3;                      // for tid<128: A row, col chunk
  const int ac = (tid & 7) << 3;
  unsigned aoff[2], boffs[2][4];
  for (int bfi = 0; bfi < 2; ++bfi) {
    aoff[bfi] = lds_off_u32(&As[bfi][0] + ar * 64 + ac);
    for (int q = 0; q < 4; ++q)
      boffs[bfi][q] = lds_off_u32(&Bs[bfi][0] + (tid + q * 256) * 8);
  }

  if (t == 0 && tid < 128) {                    // h_{-1}==0: pre-zero both bufs
    ushortx8 z = {};
    *(ushortx8*)(&As[0][0] + ar * 64 + ac) = z;
    *(ushortx8*)(&As[1][0] + ar * 64 + ac) = z;
  }

#define LSTM_ISSUE(bufi, k0v)                                                  \
  do {                                                                         \
    if (t != 0 && tid < 128)                                                   \
      async_b128(aoff[bufi], hprev + ar * HID + (k0v) + ac);                   \
    for (int q = 0; q < 4; ++q) {                                              \
      const int c8 = tid + q * 256;                                            \
      const int rr = c8 >> 3, cc = (c8 & 7) << 3;                              \
      async_b128(boffs[bufi][q],                                               \
                 Whh + (size_t)(bn * 128 + rr) * HID + (k0v) + cc);            \
    }                                                                          \
  } while (0)

  LSTM_ISSUE(0, 0);                             // prime the pipeline
  for (int k0 = 0; k0 < HID; k0 += 64) {
    const int cur = (k0 >> 6) & 1;
    if (k0) __syncthreads();                    // prev reads done before reuse
    if (k0 + 64 < HID) LSTM_ISSUE(cur ^ 1, k0 + 64);
    wait_asyncN<4>();                           // tile `cur` landed (per wave)
    __syncthreads();                            // ...for all waves
    for (int kk = 0; kk < 64; kk += 32) {
      const bf16x16 afrag = load_frag(&As[cur][0] + kk, 64);
      const bf16x16 bfrag = load_frag(&Bs[cur][0] + w * 16 * 64 + kk, 64);
      acc = wmma_bf16(afrag, bfrag, acc);
    }
  }
#undef LSTM_ISSUE

  const int lane = tid & 31;
  const int n  = lane & 15;
  const int mb = (lane >> 4) << 3;
  const int col = bn * 128 + w * 16 + n;
  const float bv = bhh[col];
  for (int r = 0; r < 8; ++r) {
    const int m = mb + r;                       // batch index
    gates[(size_t)m * GATES + col] =
        acc[r] + bv + xproj[((size_t)m * TDEC + t) * GATES + col];
  }
}

// Gate nonlinearity + state update.
__global__ __launch_bounds__(1024) void lstm_pointwise(
    const float* __restrict__ gates, float* __restrict__ c,
    unsigned short* __restrict__ hprev, float* __restrict__ hseq,
    unsigned short* __restrict__ hseq_bf,
    float* __restrict__ out_main, float* __restrict__ out_hT,
    float* __restrict__ out_cT, int t)
{
  const int b = blockIdx.x;
  const int h = threadIdx.x;
  const float gi = gates[(size_t)b * GATES + 0 * HID + h];
  const float gf = gates[(size_t)b * GATES + 1 * HID + h];
  const float gg = gates[(size_t)b * GATES + 2 * HID + h];
  const float go = gates[(size_t)b * GATES + 3 * HID + h];
  const float si = 1.0f / (1.0f + __expf(-gi));
  const float sf = 1.0f / (1.0f + __expf(-gf));
  const float tg = tanhf(gg);
  const float so = 1.0f / (1.0f + __expf(-go));
  const float cold = (t == 0) ? 0.0f : c[b * HID + h];
  const float cnew = sf * cold + si * tg;
  const float hv   = so * tanhf(cnew);
  c[b * HID + h] = cnew;
  hprev[b * HID + h] = f2bf(hv);
  hseq[((size_t)b * TDEC + t) * HID + h] = hv;
  hseq_bf[((size_t)b * TDEC + t) * HID + h] = f2bf(hv);
  out_main[((size_t)b * TDEC + t) * (3 * HID) + h] = hv;   // concat slot 0
  if (t == TDEC - 1) {
    out_hT[b * HID + h] = hv;
    out_cT[b * HID + h] = cnew;
  }
}

// ---------------------------------------------------------------------------
// Flash-style cross-attention. One block = (batch, head, 32-row q-tile).
// Q/K tiles DMA'd via async-to-LDS; S = QK^T and O += P~V on WMMA; online
// softmax with rescaled f32 accumulators; bf16 output for the O-projection.
// 2048 blocks -> latency hidden by occupancy.
// ---------------------------------------------------------------------------
__global__ __launch_bounds__(256) void attn_kernel(
    const unsigned short* __restrict__ Q,     // bf16 [B*TDEC][EMB]
    const unsigned short* __restrict__ Kenc,  // bf16 [B*TSRC][EMB]
    const unsigned short* __restrict__ Venc,  // bf16 [B*TSRC][EMB]
    const unsigned char* __restrict__ mask,   // bool [B][TSRC]
    unsigned short* __restrict__ O)           // bf16 [B*TDEC][EMB]
{
  __shared__ __attribute__((aligned(16))) unsigned short Qs[32 * 64];
  __shared__ __attribute__((aligned(16))) unsigned short Ks[128 * 64];
  __shared__ __attribute__((aligned(16))) unsigned short Vs[64 * 128];  // [dim][key]
  __shared__ __attribute__((aligned(16))) float          Sf[32 * 128];
  __shared__ __attribute__((aligned(16))) unsigned short Pb[32 * 128];
  __shared__ float rowM[32], rowL[32], rowScale[32];

  const int tid  = threadIdx.x;
  const int w    = tid >> 5;
  const int wm   = w >> 2;                    // 0..1
  const int wn   = w & 3;                     // 0..3
  const int lane = tid & 31;

  const int qt   = blockIdx.x & 7;
  const int head = (blockIdx.x >> 3) & 15;
  const int b    = blockIdx.x >> 7;

  const size_t qrow0 = (size_t)b * TDEC + qt * 32;
  const size_t krow0 = (size_t)b * TSRC;
  const int    hcol  = head * DHEAD;

  { // Q tile 32x64: async DMA, 1 chunk/thread (waited at first wait_async0)
    const int cidx = tid << 3;
    const int r = cidx >> 6, d = cidx & 63;
    async_b128(lds_off_u32(Qs + cidx), Q + (qrow0 + r) * EMB + hcol + d);
  }
  if (tid < 32) { rowM[tid] = -1e30f; rowL[tid] = 0.0f; }

  floatx8 accO = {};

  for (int kb = 0; kb < TSRC / 128; ++kb) {
    __syncthreads();
    for (int q = 0; q < 4; ++q) {             // K tile [128][64]: async DMA
      const int cidx = (tid + q * 256) << 3;
      const int r = cidx >> 6, d = cidx & 63;
      async_b128(lds_off_u32(Ks + cidx),
                 Kenc + (krow0 + kb * 128 + r) * EMB + hcol + d);
    }
    for (int i = tid; i < 8192; i += 256) {   // V tile transposed [64][128]
      const int r = i >> 6, d = i & 63;
      Vs[d * 128 + r] = Venc[(krow0 + kb * 128 + r) * EMB + hcol + d];
    }
    wait_async0();
    __syncthreads();

    // S = (Q K^T) * d^-0.5, masked
    for (int scb = 0; scb < 2; ++scb) {
      const int ct = wn + scb * 4;
      floatx8 accS = {};
      for (int kk = 0; kk < 64; kk += 32) {
        const bf16x16 af  = load_frag(Qs + wm * 16 * 64 + kk, 64);
        const bf16x16 bfr = load_frag(Ks + ct * 16 * 64 + kk, 64);
        accS = wmma_bf16(af, bfr, accS);
      }
      const int n  = lane & 15;
      const int mb = (lane >> 4) << 3;
      const int key = kb * 128 + ct * 16 + n;
      const bool mk = mask[b * TSRC + key] != 0;
      for (int r = 0; r < 8; ++r) {
        float v = accS[r] * 0.125f;           // 1/sqrt(64)
        if (mk) v = -1e9f;
        Sf[(wm * 16 + mb + r) * 128 + ct * 16 + n] = v;
      }
    }
    __syncthreads();

    // Online softmax stats: one lane per query row
    if (tid < 32) {
      const int row = tid;
      const float mold = rowM[row];
      float mnew = mold;
      for (int j = 0; j < 128; ++j) mnew = fmaxf(mnew, Sf[row * 128 + j]);
      float sum = 0.0f;
      for (int j = 0; j < 128; ++j) {
        const float p = __expf(Sf[row * 128 + j] - mnew);
        Pb[row * 128 + j] = f2bf(p);
        sum += p;
      }
      const float scl = __expf(mold - mnew);
      rowL[row] = rowL[row] * scl + sum;
      rowM[row] = mnew;
      rowScale[row] = scl;
    }
    __syncthreads();

    // O = O*scale + P~ V
    {
      const int mb = (lane >> 4) << 3;
      for (int r = 0; r < 8; ++r) accO[r] *= rowScale[wm * 16 + mb + r];
      for (int kk = 0; kk < 128; kk += 32) {
        const bf16x16 af  = load_frag(Pb + wm * 16 * 128 + kk, 128);
        const bf16x16 bfr = load_frag(Vs + wn * 16 * 128 + kk, 128);
        accO = wmma_bf16(af, bfr, accO);
      }
    }
  }
  __syncthreads();

  { // normalize, write bf16
    const int n  = lane & 15;
    const int mb = (lane >> 4) << 3;
    for (int r = 0; r < 8; ++r) {
      const int row_local = wm * 16 + mb + r;
      const float inv = 1.0f / rowL[row_local];
      O[(qrow0 + row_local) * EMB + hcol + wn * 16 + n] = f2bf(accO[r] * inv);
    }
  }
}

// ---------------------------------------------------------------------------
extern "C" void kernel_launch(void* const* d_in, const int* in_sizes, int n_in,
                              void* d_out, int out_size, void* d_ws, size_t ws_size,
                              hipStream_t stream)
{
  const float* emb   = (const float*)d_in[0];
  const float* enc_i = (const float*)d_in[1];
  const float* enc_t = (const float*)d_in[2];
  const unsigned char* mask_i = (const unsigned char*)d_in[3];
  const unsigned char* mask_t = (const unsigned char*)d_in[4];
  const float* W_ih = (const float*)d_in[5];
  const float* W_hh = (const float*)d_in[6];
  const float* b_ih = (const float*)d_in[7];
  const float* b_hh = (const float*)d_in[8];

  char* ws = (char*)d_ws;
  size_t off = 0;
  auto wsalloc = [&](size_t bytes) {
    char* p = ws + off;
    off += (bytes + 255) & ~(size_t)255;
    return p;
  };
  float*          x_proj  = (float*)wsalloc((size_t)BSZ * TDEC * GATES * 4);
  float*          h_seq   = (float*)wsalloc((size_t)BSZ * TDEC * HID * 4);
  float*          gates   = (float*)wsalloc((size_t)BSZ * GATES * 4);
  float*          cstate  = (float*)wsalloc((size_t)BSZ * HID * 4);
  unsigned short* hprev   = (unsigned short*)wsalloc((size_t)BSZ * HID * 2);
  unsigned short* hseq_bf = (unsigned short*)wsalloc((size_t)BSZ * TDEC * HID * 2);
  unsigned short* emb_bf  = (unsigned short*)wsalloc((size_t)BSZ * TDEC * EMB * 2);
  unsigned short* Wih_bf  = (unsigned short*)wsalloc((size_t)GATES * EMB * 2);
  unsigned short* Whh_bf  = (unsigned short*)wsalloc((size_t)GATES * HID * 2);
  unsigned short* enc_bf  = (unsigned short*)wsalloc((size_t)BSZ * TSRC * EMB * 2);
  unsigned short* Wq_bf   = (unsigned short*)wsalloc((size_t)EMB * EMB * 2);
  unsigned short* Wk_bf   = (unsigned short*)wsalloc((size_t)EMB * EMB * 2);
  unsigned short* Wv_bf   = (unsigned short*)wsalloc((size_t)EMB * EMB * 2);
  unsigned short* Wo_bf   = (unsigned short*)wsalloc((size_t)EMB * EMB * 2);
  unsigned short* Qb      = (unsigned short*)wsalloc((size_t)BSZ * TDEC * EMB * 2);
  unsigned short* Kb      = (unsigned short*)wsalloc((size_t)BSZ * TSRC * EMB * 2);
  unsigned short* Vb      = (unsigned short*)wsalloc((size_t)BSZ * TSRC * EMB * 2);
  unsigned short* att_bf  = (unsigned short*)wsalloc((size_t)BSZ * TDEC * EMB * 2);

  float* out    = (float*)d_out;
  float* out_hT = out + (size_t)BSZ * TDEC * 3 * HID;
  float* out_cT = out_hT + (size_t)BSZ * HID;

  const int CVT = 256;                        // cvt threads; 2048 elems/block
  // 0) one-shot bf16 conversions of LSTM operands
  cvt_f32_bf16<<<(BSZ * TDEC * EMB) / 2048, CVT, 0, stream>>>(emb, emb_bf);
  cvt_f32_bf16<<<(GATES * EMB) / 2048, CVT, 0, stream>>>(W_ih, Wih_bf);
  cvt_f32_bf16<<<(GATES * HID) / 2048, CVT, 0, stream>>>(W_hh, Whh_bf);

  // 1) x_proj = input @ W_ih^T + b_ih
  gemm_bf<<<dim3(GATES / 256, (BSZ * TDEC) / 32), 256, 0, stream>>>(
      emb_bf, Wih_bf, b_ih, nullptr, 0, x_proj, GATES, 0, EMB);

  // 2) sequential recurrence
  for (int t = 0; t < TDEC; ++t) {
    lstm_step_gemm<<<GATES / 128, 256, 0, stream>>>(hprev, Whh_bf, b_hh, x_proj,
                                                    t, gates);
    lstm_pointwise<<<BSZ, HID, 0, stream>>>(gates, cstate, hprev, h_seq, hseq_bf,
                                            out, out_hT, out_cT, t);
  }

  // 3) two cross-attentions (item, title)
  for (int tag = 0; tag < 2; ++tag) {
    const float* enc = tag ? enc_t : enc_i;
    const unsigned char* mask = tag ? mask_t : mask_i;
    const int base = 9 + tag * 8;
    const float* Wq = (const float*)d_in[base + 0];
    const float* bq = (const float*)d_in[base + 1];
    const float* Wk = (const float*)d_in[base + 2];
    const float* bk = (const float*)d_in[base + 3];
    const float* Wv = (const float*)d_in[base + 4];
    const float* bv = (const float*)d_in[base + 5];
    const float* Wo = (const float*)d_in[base + 6];
    const float* bo = (const float*)d_in[base + 7];

    cvt_f32_bf16<<<(BSZ * TSRC * EMB) / 2048, CVT, 0, stream>>>(enc, enc_bf);
    cvt_f32_bf16<<<(EMB * EMB) / 2048, CVT, 0, stream>>>(Wq, Wq_bf);
    cvt_f32_bf16<<<(EMB * EMB) / 2048, CVT, 0, stream>>>(Wk, Wk_bf);
    cvt_f32_bf16<<<(EMB * EMB) / 2048, CVT, 0, stream>>>(Wv, Wv_bf);
    cvt_f32_bf16<<<(EMB * EMB) / 2048, CVT, 0, stream>>>(Wo, Wo_bf);

    gemm_bf<<<dim3(EMB / 256, (BSZ * TDEC) / 32), 256, 0, stream>>>(
        hseq_bf, Wq_bf, bq, nullptr, 0, Qb, EMB, 1, EMB);
    gemm_bf<<<dim3(EMB / 256, (BSZ * TSRC) / 32), 256, 0, stream>>>(
        enc_bf, Wk_bf, bk, nullptr, 0, Kb, EMB, 1, EMB);
    gemm_bf<<<dim3(EMB / 256, (BSZ * TSRC) / 32), 256, 0, stream>>>(
        enc_bf, Wv_bf, bv, nullptr, 0, Vb, EMB, 1, EMB);

    attn_kernel<<<BSZ * NHEAD * (TDEC / 32), 256, 0, stream>>>(Qb, Kb, Vb, mask,
                                                               att_bf);

    // ctx = h_seq + attn @ Wo^T + bo, written into concat slot (1 or 2)
    gemm_bf<<<dim3(EMB / 256, (BSZ * TDEC) / 32), 256, 0, stream>>>(
        att_bf, Wo_bf, bo, h_seq, HID, out + HID + (size_t)tag * HID, 3 * HID, 0,
        EMB);
  }

  (void)in_sizes; (void)n_in; (void)out_size; (void)ws_size;
}
